// QRNNModel_584115553046
// MI455X (gfx1250) — matrix operand order repaired and checked by reading, not compiled, not run on hardware
//
#include <hip/hip_runtime.h>
#include <hip/hip_bf16.h>
#include <math.h>

// ---------------- problem constants ----------------
#define BATCH   8
#define SEQ     512
#define EMBD    64
#define HIDDEN  512
#define VOCABN  32000
#define G3H     (3 * HIDDEN)   // 1536
#define MROWS   (SEQ * BATCH)  // 4096

// ---------------- vector types ----------------
typedef __attribute__((ext_vector_type(16))) __bf16 v16bf;
typedef __attribute__((ext_vector_type(8)))  __bf16 v8bf;
typedef __attribute__((ext_vector_type(8)))  float  v8f;
typedef __attribute__((ext_vector_type(4)))  unsigned int v4u;
typedef __attribute__((ext_vector_type(4)))  int v4i;

union BF16Frag {
    v16bf v;
    v8bf  h[2];
};

// ---------------- CDNA5 async global->LDS copy (ASYNCcnt path) ----------------
#if defined(__gfx1250__) && __has_builtin(__builtin_amdgcn_global_load_async_to_lds_b128)
#define HAVE_ASYNC_LDS 1
#else
#define HAVE_ASYNC_LDS 0
#endif

#if HAVE_ASYNC_LDS
typedef v4i __attribute__((address_space(1)))* gas_v4i_ptr;
typedef v4i __attribute__((address_space(3)))* las_v4i_ptr;
#endif

__device__ __forceinline__ void async_copy16(const __bf16* g, __bf16* l) {
#if HAVE_ASYNC_LDS
    __builtin_amdgcn_global_load_async_to_lds_b128(
        (gas_v4i_ptr)(void*)(g),
        (las_v4i_ptr)(void*)(l),
        0, 0);
#else
    *(v4u*)l = *(const v4u*)g;
#endif
}

__device__ __forceinline__ void async_wait_all() {
#if HAVE_ASYNC_LDS
#if __has_builtin(__builtin_amdgcn_s_wait_asynccnt)
    __builtin_amdgcn_s_wait_asynccnt(0);
#else
    asm volatile("s_wait_asynccnt 0x0" ::: "memory");
#endif
#endif
}

// ---------------- weight transpose + f32->bf16 convert ----------------
// W: [K, N] row-major f32  ->  Wt: [N, K] row-major bf16
__global__ __launch_bounds__(256) void transpose_cvt_kernel(
    const float* __restrict__ W, __bf16* __restrict__ Wt, int K, int N)
{
    __shared__ float tile[32][33];
    const int kBase = blockIdx.y * 32;
    const int nBase = blockIdx.x * 32;
    const int tx = threadIdx.x;   // 0..31
    const int ty = threadIdx.y;   // 0..7
    #pragma unroll
    for (int i = 0; i < 32; i += 8) {
        int k = kBase + ty + i, n = nBase + tx;
        tile[ty + i][tx] = (k < K && n < N) ? W[(size_t)k * N + n] : 0.0f;
    }
    __syncthreads();
    #pragma unroll
    for (int i = 0; i < 32; i += 8) {
        int n = nBase + ty + i, k = kBase + tx;
        if (n < N && k < K)
            Wt[(size_t)n * K + k] = (__bf16)tile[tx][ty + i];
    }
}

// ---------------- embedding gather -> bf16 A matrix [T*B, EMBD] ----------------
__global__ __launch_bounds__(256) void embed_gather_kernel(
    const int* __restrict__ inputs, const float* __restrict__ emb,
    __bf16* __restrict__ X0)
{
    int idx = blockIdx.x * 256 + threadIdx.x;   // 0 .. T*B*EMBD-1 (exact)
    int e = idx & (EMBD - 1);
    int m = idx >> 6;            // t*B + b
    int t = m >> 3;
    int b = m & (BATCH - 1);
    int tok = inputs[b * SEQ + t];
    X0[idx] = (__bf16)emb[(size_t)tok * EMBD + e];
}

// ---------------- bf16 WMMA GEMM: C[M,N] = A[M,K] * Bt[N,K]^T + bias ----------------
// Block tile 64x64, 4 waves (wave32), each wave owns a 16-row strip x 64 cols.
// Double-buffered LDS, async global->LDS staging.
#define BM 64
#define BN 64
#define BK 32

// Each thread copies exactly 2 A-chunks and 2 B-chunks (16B each), branch-free.
__device__ __forceinline__ void stage_tile(
    const __bf16* __restrict__ A, const __bf16* __restrict__ Bt,
    int K, int m0, int n0, int k0,
    __bf16* ldsA, __bf16* ldsB, int tid)
{
#if HAVE_ASYNC_LDS
    #pragma unroll
    for (int i = 0; i < 2; ++i) {
        int c = tid + i * 128;           // 0..255, no condition needed
        int row = c >> 2, seg = c & 3;
        async_copy16(A + (size_t)(m0 + row) * K + k0 + seg * 8,
                     ldsA + row * BK + seg * 8);
    }
    #pragma unroll
    for (int i = 0; i < 2; ++i) {
        int c = tid + i * 128;
        int row = c >> 2, seg = c & 3;
        async_copy16(Bt + (size_t)(n0 + row) * K + k0 + seg * 8,
                     ldsB + row * BK + seg * 8);
    }
#else
    // Fallback: batch the 4 global loads, then the 4 LDS stores.
    int r0 = tid >> 2, s0 = tid & 3;
    int c1 = tid + 128;
    int r1 = c1 >> 2, s1 = c1 & 3;
    v4u a0 = *(const v4u*)(A  + (size_t)(m0 + r0) * K + k0 + s0 * 8);
    v4u a1 = *(const v4u*)(A  + (size_t)(m0 + r1) * K + k0 + s1 * 8);
    v4u b0 = *(const v4u*)(Bt + (size_t)(n0 + r0) * K + k0 + s0 * 8);
    v4u b1 = *(const v4u*)(Bt + (size_t)(n0 + r1) * K + k0 + s1 * 8);
    *(v4u*)(ldsA + r0 * BK + s0 * 8) = a0;
    *(v4u*)(ldsA + r1 * BK + s1 * 8) = a1;
    *(v4u*)(ldsB + r0 * BK + s0 * 8) = b0;
    *(v4u*)(ldsB + r1 * BK + s1 * 8) = b1;
#endif
}

__global__ __launch_bounds__(128) void gemm_bf16_kernel(
    const __bf16* __restrict__ A,    // [M,K] row-major
    const __bf16* __restrict__ Bt,   // [N,K] row-major (B transposed)
    const float*  __restrict__ bias, // [N]
    float* __restrict__ C,           // [M,N] row-major
    int M, int N, int K)
{
    __shared__ __align__(16) __bf16 ldsA[2][BM * BK];  // 2 x 4 KB
    __shared__ __align__(16) __bf16 ldsB[2][BN * BK];  // 2 x 4 KB

    const int tid  = threadIdx.x;
    const int lane = tid & 31;
    const int wave = tid >> 5;          // 0..3
    const int m0 = blockIdx.y * BM;
    const int n0 = blockIdx.x * BN;

    const int laneRow = lane & 15;      // 0..15
    const int laneHi  = lane >> 4;      // 0 or 1

    v8f acc[4];
    #pragma unroll
    for (int i = 0; i < 4; ++i)
        #pragma unroll
        for (int r = 0; r < 8; ++r) acc[i][r] = 0.0f;

    // prologue: stage first K-tile into buffer 0
    stage_tile(A, Bt, K, m0, n0, 0, ldsA[0], ldsB[0], tid);
    async_wait_all();
    __syncthreads();

    int cur = 0;
    for (int k0 = 0; k0 < K; k0 += BK) {
        // issue async staging of the next K-tile into the other buffer
        if (k0 + BK < K)
            stage_tile(A, Bt, K, m0, n0, k0 + BK,
                       ldsA[cur ^ 1], ldsB[cur ^ 1], tid);

        // A fragment (16x32, ISA layout): lane holds M=laneRow,
        // K in {8*laneHi .. +7} (elems 0..7) and {16+8*laneHi .. +7} (elems 8..15).
        BF16Frag a;
        {
            const __bf16* p = ldsA[cur] + (wave * 16 + laneRow) * BK + laneHi * 8;
            a.h[0] = *(const v8bf*)(p);
            a.h[1] = *(const v8bf*)(p + 16);
        }
        // Preload ALL B fragments (32x16 each: lane holds column N=laneRow,
        // K in {8*laneHi .. +7} and {16+8*laneHi .. +7}), so the dscnt wait is taken once and the
        // 4 WMMAs issue back-to-back.
        BF16Frag bfr[4];
        #pragma unroll
        for (int ns = 0; ns < 4; ++ns) {
            const __bf16* p = ldsB[cur] + (ns * 16 + laneRow) * BK + laneHi * 8;
            bfr[ns].h[0] = *(const v8bf*)(p);
            bfr[ns].h[1] = *(const v8bf*)(p + 16);
        }
        #pragma unroll
        for (int ns = 0; ns < 4; ++ns) {
            acc[ns] = __builtin_amdgcn_wmma_f32_16x16x32_bf16(
                false, a.v, false, bfr[ns].v, (short)0, acc[ns], false, false);
        }

        async_wait_all();   // own async copies into next buffer complete
        __syncthreads();    // everyone done: safe to read next buffer / reuse this one
        cur ^= 1;
    }

    // Epilogue: C/D layout lane -> N=laneRow, M rows = 8*laneHi + r
    const int mBase = m0 + wave * 16 + laneHi * 8;
    #pragma unroll
    for (int ns = 0; ns < 4; ++ns) {
        int nn = n0 + ns * 16 + laneRow;
        float bv = bias[nn];
        #pragma unroll
        for (int r = 0; r < 8; ++r)
            C[(size_t)(mBase + r) * N + nn] = acc[ns][r] + bv;
    }
}

// ---------------- QRNN fo-pool scan (sequential over T, parallel over B*H) ----
// g: [T, B, 3H] f32 pre-activations; out bf16 at out[t*strideT + b*strideB + h]
__global__ __launch_bounds__(256) void scan_kernel(
    const float* __restrict__ g, __bf16* __restrict__ out,
    int strideT, int strideB)
{
    int idx = blockIdx.x * 256 + threadIdx.x;   // 0..B*H-1 (4096)
    int b = idx >> 9;           // / HIDDEN
    int h = idx & (HIDDEN - 1);
    float c = 0.0f;
    for (int t = 0; t < SEQ; ++t) {
        const float* gp = g + ((size_t)t * BATCH + b) * G3H;
        float z = tanhf(gp[h]);
        float f = 1.0f / (1.0f + __expf(-gp[HIDDEN + h]));
        float o = 1.0f / (1.0f + __expf(-gp[2 * HIDDEN + h]));
        c = f * c + (1.0f - f) * z;
        out[(size_t)t * strideT + (size_t)b * strideB + h] = (__bf16)(o * c);
    }
}

// ---------------- launcher ----------------
extern "C" void kernel_launch(void* const* d_in, const int* in_sizes, int n_in,
                              void* d_out, int out_size, void* d_ws, size_t ws_size,
                              hipStream_t stream) {
    const int*   inputs = (const int*)  d_in[0];
    const float* emb    = (const float*)d_in[1];
    const float* W1     = (const float*)d_in[2];
    const float* b1     = (const float*)d_in[3];
    const float* W2     = (const float*)d_in[4];
    const float* b2     = (const float*)d_in[5];
    const float* Wp     = (const float*)d_in[6];
    const float* bp     = (const float*)d_in[7];
    float* out = (float*)d_out;

    // workspace partition (all offsets 256B-aligned)
    char* ws = (char*)d_ws;
    size_t off = 0;
    __bf16* Wt1 = (__bf16*)(ws + off); off += (size_t)G3H * EMBD   * 2;   // 196,608
    __bf16* Wt2 = (__bf16*)(ws + off); off += (size_t)G3H * HIDDEN * 2;   // 1,572,864
    __bf16* Wpt = (__bf16*)(ws + off); off += (size_t)VOCABN * HIDDEN * 2;// 32,768,000
    __bf16* X0  = (__bf16*)(ws + off); off += (size_t)MROWS * EMBD * 2;   // 524,288
    float*  G   = (float*) (ws + off); off += (size_t)MROWS * G3H * 4;    // 25,165,824
    __bf16* H1  = (__bf16*)(ws + off); off += (size_t)MROWS * HIDDEN * 2; // 4,194,304
    __bf16* H2  = (__bf16*)(ws + off); off += (size_t)MROWS * HIDDEN * 2; // 4,194,304

    dim3 tb(32, 8);
    // W1: [64,1536] -> Wt1 [1536,64]
    transpose_cvt_kernel<<<dim3(G3H / 32, EMBD / 32), tb, 0, stream>>>(W1, Wt1, EMBD, G3H);
    // W2: [512,1536] -> Wt2 [1536,512]
    transpose_cvt_kernel<<<dim3(G3H / 32, HIDDEN / 32), tb, 0, stream>>>(W2, Wt2, HIDDEN, G3H);
    // Wp: [512,32000] -> Wpt [32000,512]
    transpose_cvt_kernel<<<dim3(VOCABN / 32, HIDDEN / 32), tb, 0, stream>>>(Wp, Wpt, HIDDEN, VOCABN);

    // embedding gather: T*B*EMBD = 262144 threads
    embed_gather_kernel<<<(MROWS * EMBD) / 256, 256, 0, stream>>>(inputs, emb, X0);

    // layer 1 GEMM: [4096,64] x [64,1536]
    gemm_bf16_kernel<<<dim3(G3H / BN, MROWS / BM), 128, 0, stream>>>(
        X0, Wt1, b1, G, MROWS, G3H, EMBD);
    // scan 1 -> H1 rows (t*B+b)
    scan_kernel<<<(BATCH * HIDDEN) / 256, 256, 0, stream>>>(
        G, H1, BATCH * HIDDEN, HIDDEN);

    // layer 2 GEMM: [4096,512] x [512,1536]
    gemm_bf16_kernel<<<dim3(G3H / BN, MROWS / BM), 128, 0, stream>>>(
        H1, Wt2, b2, G, MROWS, G3H, HIDDEN);
    // scan 2 -> H2 rows (b*T+t)
    scan_kernel<<<(BATCH * HIDDEN) / 256, 256, 0, stream>>>(
        G, H2, HIDDEN, SEQ * HIDDEN);

    // projection GEMM: [4096,512] x [512,32000] -> logits [B,T,V]
    gemm_bf16_kernel<<<dim3(VOCABN / BN, MROWS / BM), 128, 0, stream>>>(
        H2, Wpt, bp, out, MROWS, VOCABN, HIDDEN);
}
